// LSTM_35132832481781
// MI455X (gfx1250) — compile-verified, matrix operand order checked by reading, not run on hardware
//
#include <hip/hip_runtime.h>

// Problem constants (from reference): B=4096, IN=128, H=1024, O=64, T=25
#define BDIM   4096
#define INF    128
#define HID    1024
#define OUTD   64
#define TSTEPS 25
#define K0     (INF + OUTD + HID)   // 1216 = layer-1 concat K  (x | y_prev | h1)
#define K1     (2 * HID)            // 2048 = layer-2 concat K  (h1 | h2)
#define G4H    (4 * HID)            // 4096 gate width

typedef _Float16 v16h __attribute__((ext_vector_type(16)));
typedef _Float16 v8h  __attribute__((ext_vector_type(8)));
typedef float    v8f  __attribute__((ext_vector_type(8)));

// Load one WMMA 16x16x32 f16 fragment for this lane from row-major f16 data.
// Per ISA layout: lane holds row/col (lane&15); K elements {ks..ks+7, ks+16..ks+23}
// relative to the 32-wide k-tile, ks = (lane>=16 ? 8 : 0). base already includes
// row*ld + ktile*32 + ks, so the two halves are base[0..7] and base[16..23].
__device__ __forceinline__ v16h load_frag(const _Float16* __restrict__ p) {
    union { v16h v; v8h h[2]; } u;
    u.h[0] = *(const v8h*)(p);
    u.h[1] = *(const v8h*)(p + 16);
    return u.v;
}

__device__ __forceinline__ float sigmoidf_fast(float x) {
    return 1.0f / (1.0f + __expf(-x));
}
__device__ __forceinline__ float tanhf_fast(float x) {
    float e = __expf(-2.0f * fabsf(x));
    float t = (1.0f - e) / (1.0f + e);
    return copysignf(t, x);
}

// Fused GEMM + LSTM cell.
// Gates = X[B,K] @ W[4096,K]^T + bias; gate order (PyTorch): i,f,g,o stacked in rows.
// Grid: B/32 blocks (32-row strips). Block: 256 threads = 8 waves.
// Wave w owns h-columns [w*128, (w+1)*128) as 8 tiles of 16. For each h-tile it
// keeps 8 accumulators live (2 M-tiles x 4 gates): per k-tile it loads 2 A frags
// + 4 B frags and issues 8 WMMAs (B reused 2x, A reused 4x in registers).
template <int K>
__global__ __launch_bounds__(256) void lstm_layer_kernel(
    const _Float16* __restrict__ X,     // [B, K] f16 row-major
    const _Float16* __restrict__ W,     // [4H, K] f16 row-major
    const float*    __restrict__ bias,  // [4H] f32 (b_ih + b_hh)
    float*          __restrict__ c,     // [B, H] f32 cell state (in/out)
    _Float16*       __restrict__ h_dst0, int ld0,   // h output slot 0
    _Float16*       __restrict__ h_dst1, int ld1)   // optional slot 1 (may be null)
{
    const int m0   = blockIdx.x << 5;          // 32-row batch strip
    const int wave = threadIdx.x >> 5;         // 0..7
    const int lane = threadIdx.x & 31;
    const int rc   = lane & 15;                // fragment row (A) / col (B) index
    const int ks   = (lane >> 4) << 3;         // 0 or 8
    constexpr int KT = K >> 5;                 // 32-wide k-tiles

    const _Float16* Abase0 = X + (size_t)(m0 + rc) * K + ks;
    const _Float16* Abase1 = Abase0 + (size_t)16 * K;

    for (int i = 0; i < 8; ++i) {
        const int n0 = ((wave << 3) + i) << 4;          // h-column tile base, [0,1024)
        const size_t rb = (size_t)(n0 + rc) * K + ks;
        const _Float16* Bi = W + rb;                    // i-gate rows [0,H)
        const _Float16* Bf = W + (size_t)HID     * K + rb;
        const _Float16* Bg = W + (size_t)2 * HID * K + rb;
        const _Float16* Bo = W + (size_t)3 * HID * K + rb;

        v8f a0i = {}, a0f = {}, a0g = {}, a0o = {};
        v8f a1i = {}, a1f = {}, a1g = {}, a1o = {};
        for (int kt = 0; kt < KT; ++kt) {
            const int ko = kt << 5;
            v16h a0 = load_frag(Abase0 + ko);
            v16h a1 = load_frag(Abase1 + ko);
            v16h bi = load_frag(Bi + ko);
            v16h bf = load_frag(Bf + ko);
            v16h bg = load_frag(Bg + ko);
            v16h bo = load_frag(Bo + ko);
            a0i = __builtin_amdgcn_wmma_f32_16x16x32_f16(false, a0, false, bi, (short)0, a0i, false, false);
            a1i = __builtin_amdgcn_wmma_f32_16x16x32_f16(false, a1, false, bi, (short)0, a1i, false, false);
            a0f = __builtin_amdgcn_wmma_f32_16x16x32_f16(false, a0, false, bf, (short)0, a0f, false, false);
            a1f = __builtin_amdgcn_wmma_f32_16x16x32_f16(false, a1, false, bf, (short)0, a1f, false, false);
            a0g = __builtin_amdgcn_wmma_f32_16x16x32_f16(false, a0, false, bg, (short)0, a0g, false, false);
            a1g = __builtin_amdgcn_wmma_f32_16x16x32_f16(false, a1, false, bg, (short)0, a1g, false, false);
            a0o = __builtin_amdgcn_wmma_f32_16x16x32_f16(false, a0, false, bo, (short)0, a0o, false, false);
            a1o = __builtin_amdgcn_wmma_f32_16x16x32_f16(false, a1, false, bo, (short)0, a1o, false, false);
        }

        // Elementwise LSTM cell for the two 16x16 h-tiles.
        // C/D layout: vgpr r, lanes 0-15 -> M=r; lanes 16-31 -> M=r+8; N = lane&15.
        const int ncol = n0 + rc;
        const float b_i = bias[ncol];
        const float b_f = bias[HID     + ncol];
        const float b_g = bias[2 * HID + ncol];
        const float b_o = bias[3 * HID + ncol];

        #pragma unroll
        for (int mt = 0; mt < 2; ++mt) {
            const v8f& vi = mt ? a1i : a0i;
            const v8f& vf = mt ? a1f : a0f;
            const v8f& vg = mt ? a1g : a0g;
            const v8f& vo = mt ? a1o : a0o;
            const int mbase = m0 + (mt << 4) + ((lane >> 4) << 3);
            #pragma unroll
            for (int r = 0; r < 8; ++r) {
                const int m = mbase + r;
                const size_t cidx = (size_t)m * HID + ncol;
                float gi = sigmoidf_fast(vi[r] + b_i);
                float gf = sigmoidf_fast(vf[r] + b_f);
                float gg = tanhf_fast  (vg[r] + b_g);
                float go = sigmoidf_fast(vo[r] + b_o);
                float cn = gf * c[cidx] + gi * gg;
                c[cidx] = cn;
                _Float16 hh = (_Float16)(go * tanhf_fast(cn));
                h_dst0[(size_t)m * ld0 + ncol] = hh;
                if (h_dst1) h_dst1[(size_t)m * ld1 + ncol] = hh;
            }
        }
    }
}

// y = h2 @ W_out^T + b_out. Writes f32 into d_out[:, t, :] and f16 into the
// y_prev slot of the layer-1 concatenated input (feedback for next step).
// Grid: B/32 blocks; 128 threads = 4 waves, wave w owns output cols [w*16,(w+1)*16)
// for two 16-row M-tiles.
__global__ __launch_bounds__(128) void out_gemm_kernel(
    const _Float16* __restrict__ H2, int ldh,   // [B, ldh] f16 (h2 slice)
    const _Float16* __restrict__ Wo,            // [64, 1024] f16 row-major
    const float*    __restrict__ bo,            // [64] f32
    float*          __restrict__ Y, int ldy,    // f32 out, row stride T*O
    _Float16*       __restrict__ Yp, int ldyp)  // f16 y_prev slot
{
    const int m0   = blockIdx.x << 5;
    const int wave = threadIdx.x >> 5;          // 0..3 -> N tiles over 64 cols
    const int lane = threadIdx.x & 31;
    const int rc   = lane & 15;
    const int ks   = (lane >> 4) << 3;
    const int n0   = wave << 4;

    const _Float16* A0 = H2 + (size_t)(m0 + rc) * ldh + ks;
    const _Float16* A1 = A0 + (size_t)16 * ldh;
    const _Float16* Bw = Wo + (size_t)(n0 + rc) * HID + ks;

    v8f acc0 = {}, acc1 = {};
    for (int kt = 0; kt < (HID >> 5); ++kt) {
        const int ko = kt << 5;
        v16h a0 = load_frag(A0 + ko);
        v16h a1 = load_frag(A1 + ko);
        v16h b  = load_frag(Bw + ko);
        acc0 = __builtin_amdgcn_wmma_f32_16x16x32_f16(false, a0, false, b, (short)0, acc0, false, false);
        acc1 = __builtin_amdgcn_wmma_f32_16x16x32_f16(false, a1, false, b, (short)0, acc1, false, false);
    }

    const int ncol = n0 + rc;
    const float bb = bo[ncol];
    #pragma unroll
    for (int mt = 0; mt < 2; ++mt) {
        const v8f& acc = mt ? acc1 : acc0;
        const int mbase = m0 + (mt << 4) + ((lane >> 4) << 3);
        #pragma unroll
        for (int r = 0; r < 8; ++r) {
            const int m = mbase + r;
            float y = acc[r] + bb;
            Y [(size_t)m * ldy  + ncol] = y;
            Yp[(size_t)m * ldyp + ncol] = (_Float16)y;
        }
    }
}

// dst[r][0:Ka] = f16(A[r][:]); dst[r][Ka:Ka+Kb] = f16(Bm[r][:])
__global__ void concat_convert_kernel(_Float16* __restrict__ dst,
                                      const float* __restrict__ A, int Ka,
                                      const float* __restrict__ Bm, int Kb, int rows) {
    const int K = Ka + Kb;
    const size_t total = (size_t)rows * K;
    for (size_t idx = (size_t)blockIdx.x * blockDim.x + threadIdx.x; idx < total;
         idx += (size_t)gridDim.x * blockDim.x) {
        int r = (int)(idx / K), k = (int)(idx - (size_t)r * K);
        float v = (k < Ka) ? A[(size_t)r * Ka + k] : Bm[(size_t)r * Kb + (k - Ka)];
        dst[idx] = (_Float16)v;
    }
}

__global__ void bias_add_kernel(float* __restrict__ d, const float* __restrict__ a,
                                const float* __restrict__ b, int n) {
    int i = blockIdx.x * blockDim.x + threadIdx.x;
    if (i < n) d[i] = a[i] + b[i];
}

// f32 [rows, cols] -> f16 strided destination (static input into Xcat0[:, 0:128])
__global__ void convert_strided_kernel(_Float16* __restrict__ dst, int ldd,
                                       const float* __restrict__ src, int lds,
                                       int rows, int cols) {
    const size_t total = (size_t)rows * cols;
    for (size_t idx = (size_t)blockIdx.x * blockDim.x + threadIdx.x; idx < total;
         idx += (size_t)gridDim.x * blockDim.x) {
        int r = (int)(idx / cols), k = (int)(idx - (size_t)r * cols);
        dst[(size_t)r * ldd + k] = (_Float16)src[(size_t)r * lds + k];
    }
}

extern "C" void kernel_launch(void* const* d_in, const int* in_sizes, int n_in,
                              void* d_out, int out_size, void* d_ws, size_t ws_size,
                              hipStream_t stream) {
    const float* static_in = (const float*)d_in[0];
    const float* W_ih0 = (const float*)d_in[1];
    const float* W_hh0 = (const float*)d_in[2];
    const float* b_ih0 = (const float*)d_in[3];
    const float* b_hh0 = (const float*)d_in[4];
    const float* W_ih1 = (const float*)d_in[5];
    const float* W_hh1 = (const float*)d_in[6];
    const float* b_ih1 = (const float*)d_in[7];
    const float* b_hh1 = (const float*)d_in[8];
    const float* W_out = (const float*)d_in[9];
    const float* b_out = (const float*)d_in[10];
    float* out = (float*)d_out;

    // Workspace carve-up (256B aligned)
    char* ws = (char*)d_ws;
    size_t off = 0;
    auto carve = [&](size_t bytes) {
        char* p = ws + off;
        off = (off + bytes + 255) & ~(size_t)255;
        return p;
    };
    _Float16* Wcat0 = (_Float16*)carve((size_t)G4H * K0 * sizeof(_Float16)); // [4H,1216]
    _Float16* Wcat1 = (_Float16*)carve((size_t)G4H * K1 * sizeof(_Float16)); // [4H,2048]
    _Float16* Wo    = (_Float16*)carve((size_t)OUTD * HID * sizeof(_Float16));
    float*    bcat0 = (float*)   carve((size_t)G4H * sizeof(float));
    float*    bcat1 = (float*)   carve((size_t)G4H * sizeof(float));
    _Float16* Xcat0 = (_Float16*)carve((size_t)BDIM * K0 * sizeof(_Float16)); // x|y|h1
    _Float16* Xcat1 = (_Float16*)carve((size_t)BDIM * K1 * sizeof(_Float16)); // h1|h2
    float*    c1    = (float*)   carve((size_t)BDIM * HID * sizeof(float));
    float*    c2    = (float*)   carve((size_t)BDIM * HID * sizeof(float));

    // Re-initialize all recurrent state every call (deterministic / graph-safe)
    hipMemsetAsync(Xcat0, 0, (size_t)BDIM * K0 * sizeof(_Float16), stream);
    hipMemsetAsync(Xcat1, 0, (size_t)BDIM * K1 * sizeof(_Float16), stream);
    hipMemsetAsync(c1,    0, (size_t)BDIM * HID * sizeof(float),   stream);
    hipMemsetAsync(c2,    0, (size_t)BDIM * HID * sizeof(float),   stream);

    // One-time conversions (captured in the graph; cheap vs the 25-step loop)
    concat_convert_kernel<<<1024, 256, 0, stream>>>(Wcat0, W_ih0, INF + OUTD, W_hh0, HID, G4H);
    concat_convert_kernel<<<2048, 256, 0, stream>>>(Wcat1, W_ih1, HID,        W_hh1, HID, G4H);
    concat_convert_kernel<<<64,   256, 0, stream>>>(Wo,    W_out, HID,        nullptr, 0, OUTD);
    bias_add_kernel<<<(G4H + 255) / 256, 256, 0, stream>>>(bcat0, b_ih0, b_hh0, G4H);
    bias_add_kernel<<<(G4H + 255) / 256, 256, 0, stream>>>(bcat1, b_ih1, b_hh1, G4H);
    convert_strided_kernel<<<1024, 256, 0, stream>>>(Xcat0, K0, static_in, INF, BDIM, INF);

    for (int t = 0; t < TSTEPS; ++t) {
        // Layer 1: gates from [x|y_prev|h1]; h1 -> Xcat0 h-slot and Xcat1 slot 0
        lstm_layer_kernel<K0><<<BDIM / 32, 256, 0, stream>>>(
            Xcat0, Wcat0, bcat0, c1,
            Xcat0 + (INF + OUTD), K0, Xcat1, K1);
        // Layer 2: gates from [h1|h2]; h2 -> Xcat1 slot 1
        lstm_layer_kernel<K1><<<BDIM / 32, 256, 0, stream>>>(
            Xcat1, Wcat1, bcat1, c2,
            Xcat1 + HID, K1, nullptr, 0);
        // Output head: y -> d_out[:, t, :] (f32) and y_prev slot of Xcat0 (f16)
        out_gemm_kernel<<<BDIM / 32, 128, 0, stream>>>(
            Xcat1 + HID, K1, Wo, b_out,
            out + (size_t)t * OUTD, TSTEPS * OUTD,
            Xcat0 + INF, K0);
    }
}
